// SemanticSubspaceAttention_11871289606894
// MI455X (gfx1250) — compile-verified, matrix operand order checked by reading, not run on hardware
//
#include <hip/hip_runtime.h>

typedef __bf16 bf16_t;
typedef __attribute__((ext_vector_type(16))) __bf16 v16bf;
typedef __attribute__((ext_vector_type(8)))  float  v8f;

#define WMMA_BF16(a, b, c) \
  __builtin_amdgcn_wmma_f32_16x16x32_bf16(false, (a), false, (b), (short)0, (c), false, false)

constexpr int Bb = 2, Nn = 1024, Mm = 1024, Ss = 4, Dd = 512, Hh = 8, Cc = 64;

// ---- fragment loaders (wave32 16-bit A/B layout: lane holds row/col = lane&15,
// ---- K values: e in [0,8) -> k = half*8 + e ; e in [8,16) -> k = 16 + half*8 + (e-8))
__device__ inline v16bf frag_from_f32(const float* __restrict__ p, int half) {
  v16bf r;
#pragma unroll
  for (int g = 0; g < 2; ++g) {
    const float* src = p + g * 16 + half * 8;
#pragma unroll
    for (int j = 0; j < 8; ++j) r[g * 8 + j] = (bf16_t)src[j];
  }
  return r;
}

__device__ inline v16bf frag_from_bf16(const bf16_t* __restrict__ p, int half) {
  v16bf r;
#pragma unroll
  for (int g = 0; g < 2; ++g) {
    const bf16_t* src = p + g * 16 + half * 8;
#pragma unroll
    for (int j = 0; j < 8; ++j) r[g * 8 + j] = src[j];
  }
  return r;
}

// ---- K1: Y[b,n,j] = X[b,n,:]·W[j,:] + bias[j]  (bf16 out; compile-time transposed copy for V)
template <bool WRITE_T>
__global__ __launch_bounds__(32) void proj_kernel(
    const float* __restrict__ X, const float* __restrict__ W,
    const float* __restrict__ bias, bf16_t* __restrict__ Y,
    bf16_t* __restrict__ YT) {
  const int lane = threadIdx.x;
  const int mrow = lane & 15;
  const int half = lane >> 4;
  const int jt = blockIdx.x % (Dd / 16);
  const int nt = blockIdx.x / (Dd / 16);
  const int b  = blockIdx.y;

  const float* xrow = X + ((size_t)b * Nn + nt * 16 + mrow) * Dd;  // A row
  const float* wrow = W + (size_t)(jt * 16 + mrow) * Dd;           // B col (= W row j)

  v8f acc = {};
  for (int ks = 0; ks < Dd; ks += 32) {
    v16bf a = frag_from_f32(xrow + ks, half);
    v16bf w = frag_from_f32(wrow + ks, half);
    acc = WMMA_BF16(a, w, acc);
  }
  const int col = jt * 16 + mrow;
  const float bv = bias[col];
#pragma unroll
  for (int v = 0; v < 8; ++v) {
    const int r = nt * 16 + v + 8 * half;   // C/D layout: row = vgpr + 8*half
    const float val = acc[v] + bv;
    Y[((size_t)b * Nn + r) * Dd + col] = (bf16_t)val;
    if constexpr (WRITE_T) {  // transposed (B,H,C,M) copy for the attn@V B-operand
      const int h = col >> 6, c = col & 63;
      YT[(((size_t)b * Hh + h) * Cc + c) * Mm + r] = (bf16_t)val;
    }
  }
}

// ---- K2: dot[b,h,n,m] = q·k (shared across all s) -> written into attn s=0 slice
__global__ __launch_bounds__(32) void qk_kernel(
    const bf16_t* __restrict__ Q, const bf16_t* __restrict__ K,
    float* __restrict__ attn) {
  const int lane = threadIdx.x;
  const int mrow = lane & 15;
  const int half = lane >> 4;
  const int bh = blockIdx.y;
  const int b = bh / Hh, h = bh % Hh;
  const int nt = blockIdx.x / (Mm / 64);
  const int mg = blockIdx.x % (Mm / 64);

  const bf16_t* qrow = Q + ((size_t)b * Nn + nt * 16 + mrow) * Dd + h * Cc;
  v16bf a0 = frag_from_bf16(qrow, half);       // k = 0..31
  v16bf a1 = frag_from_bf16(qrow + 32, half);  // k = 32..63

  float* dot = attn + ((size_t)(b * Hh + h) * Ss) * Nn * Mm;  // s = 0 slice
#pragma unroll
  for (int t = 0; t < 4; ++t) {
    const int m0 = mg * 64 + t * 16;
    const bf16_t* krow = K + ((size_t)b * Mm + m0 + mrow) * Dd + h * Cc;
    v16bf b0 = frag_from_bf16(krow, half);
    v16bf b1 = frag_from_bf16(krow + 32, half);
    v8f acc = {};
    acc = WMMA_BF16(a0, b0, acc);
    acc = WMMA_BF16(a1, b1, acc);
#pragma unroll
    for (int v = 0; v < 8; ++v)
      dot[(size_t)(nt * 16 + v + 8 * half) * Mm + m0 + mrow] = acc[v];
  }
}

// ---- K3: per (b,h,n): read dot row once, emit softmax(qsub*ksub*scale*dot) for all s
__global__ __launch_bounds__(256) void softmax_kernel(
    float* __restrict__ attn, const float* __restrict__ qsub,
    const float* __restrict__ ksub) {
  const int n = blockIdx.x;
  const int bh = blockIdx.y;
  const int b = bh / Hh, h = bh % Hh;
  const int tid = threadIdx.x;
  const int wid = tid >> 5, lid = tid & 31;

  __shared__ float sred[8];
  __shared__ float sbc;

  const float* dotrow = attn + (((size_t)(b * Hh + h) * Ss) * Nn + n) * Mm;
  float d[4];
#pragma unroll
  for (int i = 0; i < 4; ++i) d[i] = dotrow[tid + i * 256];

  for (int s = 0; s < Ss; ++s) {
    const float rscale = qsub[((size_t)b * Ss + s) * Nn + n] * 0.125f;  // 1/sqrt(64)
    float t[4];
    float mx = -3.0e38f;
#pragma unroll
    for (int i = 0; i < 4; ++i) {
      t[i] = d[i] * ksub[((size_t)b * Ss + s) * Mm + tid + i * 256] * rscale;
      mx = fmaxf(mx, t[i]);
    }
#pragma unroll
    for (int off = 16; off > 0; off >>= 1) mx = fmaxf(mx, __shfl_xor(mx, off));
    if (lid == 0) sred[wid] = mx;
    __syncthreads();
    if (tid == 0) {
      float v = sred[0];
#pragma unroll
      for (int i = 1; i < 8; ++i) v = fmaxf(v, sred[i]);
      sbc = v;
    }
    __syncthreads();
    mx = sbc;

    float sum = 0.f;
#pragma unroll
    for (int i = 0; i < 4; ++i) { t[i] = __expf(t[i] - mx); sum += t[i]; }
#pragma unroll
    for (int off = 16; off > 0; off >>= 1) sum += __shfl_xor(sum, off);
    if (lid == 0) sred[wid] = sum;
    __syncthreads();
    if (tid == 0) {
      float v = 0.f;
#pragma unroll
      for (int i = 0; i < 8; ++i) v += sred[i];
      sbc = v;
    }
    __syncthreads();
    const float inv = 1.0f / sbc;

    float* out = attn + (((size_t)(b * Hh + h) * Ss + s) * Nn + n) * Mm;
#pragma unroll
    for (int i = 0; i < 4; ++i) out[tid + i * 256] = t[i] * inv;
    __syncthreads();
  }
}

// ---- K4: hidden[b,s,n,h*C+c] = attn[b,h,s,n,:]·v[b,:,h*C+c] via WMMA over K=M
__global__ __launch_bounds__(32) void av_kernel(
    const float* __restrict__ attn, const bf16_t* __restrict__ VT,
    float* __restrict__ hidden) {
  const int lane = threadIdx.x;
  const int mrow = lane & 15;
  const int half = lane >> 4;
  const int nt = blockIdx.x;
  const int h = blockIdx.y / Ss, s = blockIdx.y % Ss;
  const int b = blockIdx.z;

  const float* arow =
      attn + (((size_t)(b * Hh + h) * Ss + s) * Nn + nt * 16 + mrow) * Mm;

  v8f acc[4] = {};
  for (int ks = 0; ks < Mm; ks += 32) {
    if (ks + 32 < Mm)  // prefetch next K-step of the streamed attn row
      __builtin_prefetch(arow + ks + 32 + half * 8, 0, 0);
    v16bf a = frag_from_f32(arow + ks, half);
#pragma unroll
    for (int t = 0; t < 4; ++t) {
      const bf16_t* vrow =
          VT + ((size_t)(b * Hh + h) * Cc + t * 16 + mrow) * Mm + ks;
      v16bf bb = frag_from_bf16(vrow, half);
      acc[t] = WMMA_BF16(a, bb, acc[t]);
    }
  }
#pragma unroll
  for (int t = 0; t < 4; ++t) {
#pragma unroll
    for (int v = 0; v < 8; ++v) {
      const int r = nt * 16 + v + 8 * half;
      const int c = t * 16 + mrow;
      hidden[(((size_t)b * Ss + s) * Nn + r) * Dd + h * Cc + c] = acc[t][v];
    }
  }
}

extern "C" void kernel_launch(void* const* d_in, const int* in_sizes, int n_in,
                              void* d_out, int out_size, void* d_ws, size_t ws_size,
                              hipStream_t stream) {
  const float* in_q = (const float*)d_in[0];
  const float* in_k = (const float*)d_in[1];
  const float* in_v = (const float*)d_in[2];
  const float* qsub = (const float*)d_in[3];
  const float* ksub = (const float*)d_in[4];
  const float* Wq = (const float*)d_in[5];
  const float* bq = (const float*)d_in[6];
  const float* Wk = (const float*)d_in[7];
  const float* bk = (const float*)d_in[8];
  const float* Wv = (const float*)d_in[9];
  const float* bv = (const float*)d_in[10];

  float* hidden = (float*)d_out;                              // (B,S,N,D)
  float* attn = hidden + (size_t)Bb * Ss * Nn * Dd;           // (B,H,S,N,M)

  bf16_t* qb = (bf16_t*)d_ws;                                 // (B,N,D)
  bf16_t* kb = qb + (size_t)Bb * Nn * Dd;                     // (B,M,D)
  bf16_t* vb = kb + (size_t)Bb * Mm * Dd;                     // (B,M,D)
  bf16_t* vT = vb + (size_t)Bb * Mm * Dd;                     // (B,H,C,M)

  dim3 g1((Nn / 16) * (Dd / 16), Bb);
  proj_kernel<false><<<g1, 32, 0, stream>>>(in_q, Wq, bq, qb, nullptr);
  proj_kernel<false><<<g1, 32, 0, stream>>>(in_k, Wk, bk, kb, nullptr);
  proj_kernel<true><<<g1, 32, 0, stream>>>(in_v, Wv, bv, vb, vT);

  dim3 g2((Nn / 16) * (Mm / 64), Bb * Hh);
  qk_kernel<<<g2, 32, 0, stream>>>(qb, kb, attn);

  dim3 g3(Nn, Bb * Hh);
  softmax_kernel<<<g3, 256, 0, stream>>>(attn, qsub, ksub);

  dim3 g4(Nn / 16, Hh * Ss, Bb);
  av_kernel<<<g4, 32, 0, stream>>>(attn, vT, hidden);
}
